// Quantize_56461640073308
// MI455X (gfx1250) — matrix operand order repaired and checked by reading, not compiled, not run on hardware
//
#include <hip/hip_runtime.h>

// ---------------------------------------------------------------------------
// Fused VQ + Gumbel-softmax for gfx1250 (MI455X), TDM double-buffered.
//   prep kernel:  codebook f32 -> bf16 (row-major + transposed) + |c|^2  (ws)
//   main kernel:  per 128-token block, stream codebook in 128-code chunks via
//                 tensor_load_to_lds (TENSORcnt, double-buffered LDS);
//                 GEMM1/GEMM2 with v_wmma_f32_16x16x32_bf16, online softmax,
//                 argmin tracking.
// ---------------------------------------------------------------------------

typedef __attribute__((ext_vector_type(16))) __bf16 v16bf;
typedef __attribute__((ext_vector_type(8)))  float  v8f;
typedef __attribute__((ext_vector_type(4)))  unsigned int v4u;
typedef __attribute__((ext_vector_type(8)))  int    v8i;
typedef __attribute__((ext_vector_type(4)))  int    v4i;

#define BM       128   // tokens per block (8 waves x 16)
#define KC       128   // codebook chunk (codes)
#define DD       128   // embed dim
#define KK       2048  // codebook size
#define NTHREADS 256
#define NWAVES   8
#define NCHUNK   (KK / KC)

union FragU { v16bf v; unsigned int u[8]; };

// A-matrix fragment (16x32, 16-bit), src row-major [m][k], ld in elements.
__device__ __forceinline__ v16bf lds_frag_a(const __bf16* src, int ld) {
  const int lane = threadIdx.x & 31;
  const int m = lane & 15;
  const int hi = lane >> 4;
  const __bf16* row = src + m * ld + 8 * hi;
  FragU f;
#pragma unroll
  for (int v = 0; v < 4; ++v) f.u[v] = *(const unsigned int*)(row + 2 * v);
#pragma unroll
  for (int v = 0; v < 4; ++v) f.u[4 + v] = *(const unsigned int*)(row + 16 + 2 * v);
  return f.v;
}

// B-matrix fragment (32x16, 16-bit), src stored n-major: src[n*ld + k].
__device__ __forceinline__ v16bf lds_frag_b(const __bf16* src, int ld) {
  const int lane = threadIdx.x & 31;
  const int n = lane & 15;
  const int hi = lane >> 4;
  const __bf16* row = src + n * ld + 8 * hi;
  FragU f;
#pragma unroll
  for (int v = 0; v < 4; ++v) f.u[v] = *(const unsigned int*)(row + 2 * v);
#pragma unroll
  for (int v = 0; v < 4; ++v) f.u[4 + v] = *(const unsigned int*)(row + 16 + 2 * v);
  return f.v;
}

__device__ __forceinline__ float gumbel_noise(unsigned int token, unsigned int code) {
  unsigned int h = token * 2048u + code;
  h += 0x9E3779B9u;
  h ^= h >> 16; h *= 0x7FEB352Du;
  h ^= h >> 15; h *= 0x846CA68Bu;
  h ^= h >> 16;
  float u = (float)(h >> 8) * 5.9604645e-08f + 1.0e-9f;
  return -__logf(-__logf(u));
}

// Issue one TDM 2-D tile load (bf16 elements) global -> LDS.
// D# per CDNA5 ISA 8.3/8.4: group0 {count,lds_addr,global_addr,type=2},
// group1 {data_size=2B, tensor_dim0/1, tile_dim0/1, tensor_dim0_stride}.
// This toolchain's builtin takes 6 args: (v4u, v8i, v4i, v4i, v8i, cpol).
__device__ __forceinline__ void tdm_load_2d(unsigned lds_off, const void* gptr,
                                            unsigned tensor_d0, unsigned tensor_d1,
                                            unsigned tile_d0, unsigned tile_d1,
                                            unsigned long long stride0) {
  unsigned long long ga = (unsigned long long)gptr;
  v4u g0;
  g0[0] = 1u;                                        // count=1, user descriptor
  g0[1] = lds_off;                                   // lds_addr (bytes)
  g0[2] = (unsigned)(ga & 0xffffffffu);              // global_addr[31:0]
  g0[3] = (unsigned)((ga >> 32) & 0x01ffffffu) | (2u << 30);  // [56:32] | type=2
  v8i g1;
  g1[0] = (int)(1u << 16);                           // data_size=1 -> 2 bytes
  g1[1] = (int)((tensor_d0 & 0xffffu) << 16);        // tensor_dim0[15:0] @ [63:48]
  g1[2] = (int)((tensor_d0 >> 16) | ((tensor_d1 & 0xffffu) << 16));
  g1[3] = (int)((tensor_d1 >> 16) | ((tile_d0 & 0xffffu) << 16));
  g1[4] = (int)(tile_d1 & 0xffffu);                  // tile_dim1 (tile_dim2=0)
  g1[5] = (int)(stride0 & 0xffffffffu);              // tensor_dim0_stride lo
  g1[6] = (int)((stride0 >> 32) & 0xffffu);          // stride hi, dim1_stride=0
  g1[7] = 0;
  v4i z4 = {0, 0, 0, 0};
  v8i z8 = {0, 0, 0, 0, 0, 0, 0, 0};
  __builtin_amdgcn_tensor_load_to_lds(g0, g1, z4, z4, z8, 0);
}

// ---------------- prep: codebook f32 -> bf16 (2 layouts) + |c|^2 -----------
__global__ __launch_bounds__(NTHREADS) void vq_prep(
    const float* __restrict__ w, __bf16* __restrict__ wsC,
    __bf16* __restrict__ wsCT, float* __restrict__ wsC2) {
  const int t = threadIdx.x;
  const int k = (int)blockIdx.x * 16 + (t >> 4);   // code row
  const int d0 = (t & 15) * 8;                     // 8 dims per thread
  const float4* src = (const float4*)(w + (size_t)k * DD + d0);
  float4 a = src[0], b = src[1];
  float e[8] = {a.x, a.y, a.z, a.w, b.x, b.y, b.z, b.w};
  float ssq = 0.0f;
#pragma unroll
  for (int i = 0; i < 8; ++i) ssq += e[i] * e[i];
  __bf16* dst = wsC + (size_t)k * DD + d0;
#pragma unroll
  for (int i = 0; i < 8; ++i) dst[i] = (__bf16)e[i];
#pragma unroll
  for (int i = 0; i < 8; ++i) wsCT[(size_t)(d0 + i) * KK + k] = (__bf16)e[i];
#pragma unroll
  for (int m = 1; m < 16; m <<= 1) ssq += __shfl_xor(ssq, m, 32);
  if ((t & 15) == 0) wsC2[k] = ssq;
}

// ---------------- main fused kernel ----------------------------------------
__global__ __launch_bounds__(NTHREADS) void vq_gumbel_fused(
    const float* __restrict__ x, const __bf16* __restrict__ wsC,
    const __bf16* __restrict__ wsCT, const float* __restrict__ wsC2,
    const float* __restrict__ temp, float* __restrict__ out, int n_tokens) {
  extern __shared__ char smem_raw[];
  __bf16* sX  = (__bf16*)smem_raw;             // BM x DD                (32 KB)
  __bf16* sC  = sX + BM * DD;                  // 2 x KC x DD code-major (64 KB)
  __bf16* sCT = sC + 2 * KC * DD;              // 2 x DD x KC dim-major  (64 KB)
  __bf16* sP  = sCT + 2 * DD * KC;             // NWAVES x 16 x KC       (32 KB)

  const int tid  = threadIdx.x;
  const int lane = tid & 31;
  const int wave = tid >> 5;
  const int ncol = lane & 15;
  const int hi   = lane >> 4;
  const int m0   = (int)blockIdx.x * BM;
  const float invT = 1.0f / temp[0];

  // LDS byte offsets for TDM descriptors (generic addr low 32 bits = LDS off)
  const unsigned sC_off  = (unsigned)(unsigned long long)(void*)sC;
  const unsigned sCT_off = (unsigned)(unsigned long long)(void*)sCT;

  // kick off chunk 0 DMA (wave 0 only; EXEC ignored by tensor ops)
  if (wave == 0) {
    tdm_load_2d(sC_off, wsC, KK * DD, 1, KC * DD, 1, (unsigned long long)KK * DD);
    tdm_load_2d(sCT_off, wsCT, KK, DD, KC, DD, (unsigned long long)KK);
  }

  // ---- stage x tile -> bf16 LDS while DMA flies ----
  {
    const float4* src = (const float4*)(x + (size_t)m0 * DD);
#pragma unroll
    for (int it = 0; it < (BM * DD / 4) / NTHREADS; ++it) {
      int i = tid + it * NTHREADS;
      float4 v = src[i];
      __bf16* d = sX + i * 4;
      d[0] = (__bf16)v.x; d[1] = (__bf16)v.y; d[2] = (__bf16)v.z; d[3] = (__bf16)v.w;
    }
  }
  __syncthreads();

  v16bf ax[4];
#pragma unroll
  for (int ks = 0; ks < 4; ++ks)
    ax[ks] = lds_frag_a(sX + wave * 16 * DD + ks * 32, DD);

  v8f acc[8];
#pragma unroll
  for (int dt = 0; dt < 8; ++dt)
#pragma unroll
    for (int j = 0; j < 8; ++j) acc[dt][j] = 0.0f;

  float mrun[8], lrun[8], bestv[8];
  int besti[8];
#pragma unroll
  for (int j = 0; j < 8; ++j) {
    mrun[j] = -3.0e38f; lrun[j] = 0.0f; bestv[j] = -3.0e38f; besti[j] = 0;
  }

  for (int c = 0; c < NCHUNK; ++c) {
    const int kc0 = c * KC;
    const int cur = c & 1;
    __bf16* sCc  = sC + cur * KC * DD;
    __bf16* sCTc = sCT + cur * DD * KC;

    __syncthreads();  // all waves done with the buffer we are about to refill
    if (wave == 0) {
      if (c + 1 < NCHUNK) {   // prefetch next chunk into the other buffer
        const int nxt = 1 - cur;
        tdm_load_2d(sC_off + (unsigned)(nxt * KC * DD * 2),
                    wsC + (size_t)(kc0 + KC) * DD, KK * DD, 1, KC * DD, 1,
                    (unsigned long long)KK * DD);
        tdm_load_2d(sCT_off + (unsigned)(nxt * DD * KC * 2),
                    wsCT + (kc0 + KC), KK, DD, KC, DD, (unsigned long long)KK);
        __builtin_amdgcn_s_wait_tensorcnt(2);  // current chunk's 2 ops done
      } else {
        __builtin_amdgcn_s_wait_tensorcnt(0);
      }
    }
    __syncthreads();  // chunk c LDS data visible to all waves

    // ---- GEMM1: S(16x128) = x(16x128) . chunk^T ----
    v8f S[8];
#pragma unroll
    for (int t = 0; t < 8; ++t) {
      v8f s;
#pragma unroll
      for (int j = 0; j < 8; ++j) s[j] = 0.0f;
#pragma unroll
      for (int ks = 0; ks < 4; ++ks) {
        v16bf b = lds_frag_b(sCc + t * 16 * DD + ks * 32, DD);
        s = __builtin_amdgcn_wmma_f32_16x16x32_bf16(false, ax[ks], false, b,
                                                    (short)0, s, false, false);
      }
      S[t] = s;
    }

    // ---- logits in place; argmax of (2s - |c|^2) == argmin(dist) ----
#pragma unroll
    for (int t = 0; t < 8; ++t) {
      const int code = kc0 + t * 16 + ncol;
      const float c2 = wsC2[code];
#pragma unroll
      for (int j = 0; j < 8; ++j) {
        float raw = 2.0f * S[t][j] - c2;
        if (raw > bestv[j]) { bestv[j] = raw; besti[j] = code; }
        const int token = m0 + wave * 16 + j + 8 * hi;
        S[t][j] = (raw + gumbel_noise((unsigned)token, (unsigned)code)) * invT;
      }
    }

    // ---- online softmax ----
    float mnew[8];
    v8f av;
#pragma unroll
    for (int j = 0; j < 8; ++j) {
      float cm = S[0][j];
#pragma unroll
      for (int t = 1; t < 8; ++t) cm = fmaxf(cm, S[t][j]);
#pragma unroll
      for (int dlt = 1; dlt < 16; dlt <<= 1) cm = fmaxf(cm, __shfl_xor(cm, dlt, 32));
      mnew[j] = fmaxf(mrun[j], cm);
      float a = __expf(mrun[j] - mnew[j]);
      mrun[j] = mnew[j];
      lrun[j] *= a;
      av[j] = a;
    }
#pragma unroll
    for (int t = 0; t < 8; ++t) {
#pragma unroll
      for (int j = 0; j < 8; ++j) {
        float p = __expf(S[t][j] - mnew[j]);
        S[t][j] = p;
        sP[(wave * 16 + j + 8 * hi) * KC + t * 16 + ncol] = (__bf16)p;
      }
    }
#pragma unroll
    for (int j = 0; j < 8; ++j) {
      float rs = S[0][j];
#pragma unroll
      for (int t = 1; t < 8; ++t) rs += S[t][j];
#pragma unroll
      for (int dlt = 1; dlt < 16; dlt <<= 1) rs += __shfl_xor(rs, dlt, 32);
      lrun[j] += rs;
    }
#pragma unroll
    for (int dt = 0; dt < 8; ++dt) acc[dt] *= av;

    asm volatile("s_wait_dscnt 0" ::: "memory");  // own-wave sP stores visible

    // ---- GEMM2: acc(16x128) += P(16x128) . chunk(128x128) ----
#pragma unroll
    for (int ks = 0; ks < 4; ++ks) {
      v16bf ap = lds_frag_a(sP + wave * 16 * KC + ks * 32, KC);
#pragma unroll
      for (int dt = 0; dt < 8; ++dt) {
        v16bf b = lds_frag_b(sCTc + dt * 16 * KC + ks * 32, KC);
        acc[dt] = __builtin_amdgcn_wmma_f32_16x16x32_bf16(false, ap, false, b,
                                                          (short)0, acc[dt],
                                                          false, false);
      }
    }
  }

  // ---- epilogue: emb = acc / l ; ids via 16-lane arg-reduction ----
#pragma unroll
  for (int j = 0; j < 8; ++j) {
    const int token = m0 + wave * 16 + j + 8 * hi;
    const float inv = 1.0f / lrun[j];
#pragma unroll
    for (int dt = 0; dt < 8; ++dt)
      out[(size_t)token * DD + dt * 16 + ncol] = acc[dt][j] * inv;

    float bv = bestv[j]; int bi = besti[j];
#pragma unroll
    for (int dlt = 1; dlt < 16; dlt <<= 1) {
      float ov = __shfl_xor(bv, dlt, 32);
      int   oi = __shfl_xor(bi, dlt, 32);
      if (ov > bv || (ov == bv && oi < bi)) { bv = ov; bi = oi; }
    }
    if (ncol == 0)
      out[(size_t)n_tokens * DD + token] = (float)bi;
  }
}

extern "C" void kernel_launch(void* const* d_in, const int* in_sizes, int n_in,
                              void* d_out, int out_size, void* d_ws, size_t ws_size,
                              hipStream_t stream) {
  (void)n_in; (void)out_size; (void)ws_size;
  const float* x    = (const float*)d_in[0];
  const float* w    = (const float*)d_in[1];
  const float* temp = (const float*)d_in[2];
  float* out = (float*)d_out;
  const int n_tokens = in_sizes[0] / DD;  // 65536

  // workspace: bf16 codebook (row-major), bf16 codebook^T, |c|^2
  char* ws = (char*)d_ws;
  __bf16* wsC  = (__bf16*)ws;                            // 512 KB
  __bf16* wsCT = (__bf16*)(ws + (size_t)KK * DD * 2);    // 512 KB
  float*  wsC2 = (float*)(ws + (size_t)KK * DD * 4);     // 8 KB

  hipLaunchKernelGGL(vq_prep, dim3(KK / 16), dim3(NTHREADS), 0, stream,
                     w, wsC, wsCT, wsC2);

  // LDS: sX + 2*sC + 2*sCT + sP = 192 KB (dynamic; WGP has 320 KB)
  const size_t smem = (size_t)(BM * DD + 2 * KC * DD + 2 * DD * KC +
                               NWAVES * 16 * KC) * 2;
  hipLaunchKernelGGL(vq_gumbel_fused, dim3(n_tokens / BM), dim3(NTHREADS), smem,
                     stream, x, wsC, wsCT, wsC2, temp, out, n_tokens);
}
